// HierarchicalAttention_3358664425919
// MI455X (gfx1250) — matrix operand order repaired and checked by reading, not compiled, not run on hardware
//
#include <hip/hip_runtime.h>
#include <math.h>

typedef __attribute__((ext_vector_type(16))) __bf16 v16bf;
typedef __attribute__((ext_vector_type(8)))  __bf16 v8bf;
typedef __attribute__((ext_vector_type(8)))  float  v8f;

#define NB 512                 // batch
#define NT 512                 // seq
#define NF 512                 // features (= K of the GEMM)
#define NU 128                 // temporal units (= N of the GEMM)
#define TROWS 32               // T-tile rows per iteration
#define NTILES (NT / TROWS)    // 16
#define WSTR (NF + 8)          // bf16 Wt stride: 520 -> 1040B: 16B-aligned, 4-bank shift
#define SSTR (NF + 4)          // f32 input-tile stride: 516 -> 2064B: 16B-aligned, 4-bank shift

// LDS carve (bytes)
#define OFF_WT   0
#define SZ_WT    (NU * WSTR * 2)            // 133120
#define OFF_IN0  (OFF_WT + SZ_WT)
#define SZ_IN    (TROWS * SSTR * 4)         // 66048
#define OFF_IN1  (OFF_IN0 + SZ_IN)
#define OFF_A2   (OFF_IN1 + SZ_IN)
#define SZ_A2    (NT * 4)                   // 2048
#define OFF_PART (OFF_A2 + SZ_A2)
#define SZ_PART  (8 * 16 * 4)               // 512
#define OFF_BT   (OFF_PART + SZ_PART)
#define SZ_BT    (NU * 4)
#define OFF_UT   (OFF_BT + SZ_BT)
#define SZ_UT    (NU * 4)
#define SMEM_BYTES (OFF_UT + SZ_UT)         // 268800 B ~= 262.5 KB < 320 KB

// ---- CDNA5 async global->LDS copy (ASYNCcnt-tracked, no VGPR round-trip) ----
// Generic pointers to __shared__ carry the LDS offset in their low 32 bits
// (ISA 10.2, LDS aperture).
__device__ __forceinline__ void async_ld16(void* ldsPtr, const void* gPtr) {
    const unsigned           loff = (unsigned)(unsigned long long)ldsPtr;
    const unsigned long long ga   = (unsigned long long)gPtr;
    asm volatile("global_load_async_to_lds_b128 %0, %1, off"
                 :: "v"(loff), "v"(ga) : "memory");
}
__device__ __forceinline__ void wait_async0() {
    asm volatile("s_wait_asynccnt 0x0" ::: "memory");
}

// ---------------------------------------------------------------------------
// Fused kernel: one block per batch b, single pass over inputs (online softmax).
//   a2[t]    = tanh(inputs[b,t,:] @ Wt + bt) @ ut        (bf16 WMMA)
//   tw       = softmax_T(a2)
//   ctx[b,:] = sum_t tw[t] * inputs[b,t,:]               (f32 inputs from LDS)
//   fw       = 1.0 (softmax over a singleton axis in the reference)
// Input tiles are double-buffered in LDS via async-to-LDS DMA so the next
// tile streams from HBM while the current tile is consumed by the WMMAs.
// ---------------------------------------------------------------------------
__global__ void __launch_bounds__(256)
fused_attn_kernel(const float* __restrict__ inp,
                  const float* __restrict__ Wt,
                  const float* __restrict__ bt,
                  const float* __restrict__ ut,
                  float* __restrict__ ctx,   // [NB, NF]
                  float* __restrict__ fw,    // [NT * NF] (all ones)
                  float* __restrict__ tw)    // [NB, NT]
{
    extern __shared__ char smem[];
    __bf16* sWt  = (__bf16*)(smem + OFF_WT);   // [NU][WSTR] transposed bf16
    float*  sIn0 = (float*) (smem + OFF_IN0);  // [TROWS][SSTR] f32 tile, buf 0
    float*  sIn1 = (float*) (smem + OFF_IN1);  // [TROWS][SSTR] f32 tile, buf 1
    float*  sA2  = (float*) (smem + OFF_A2);   // [NT] logits
    float*  sPt  = (float*) (smem + OFF_PART); // [8 waves][16 rows] partials
    float*  sBt  = (float*) (smem + OFF_BT);   // [NU]
    float*  sUt  = (float*) (smem + OFF_UT);   // [NU]

    const int tid    = threadIdx.x;
    const int lane   = tid & 31;
    const int wave   = tid >> 5;
    const int laneHi = lane >> 4;
    const int lane16 = lane & 15;
    const int mt     = wave & 1;      // which 16-row M-tile of the 32-row T tile
    const int ng     = wave >> 1;     // N-group: ntiles {2*ng, 2*ng+1}
    const int b      = blockIdx.x;

    const float* gbase = inp + (long)b * NT * NF;

    // ---- prologue: kick off async DMA of tile 0 into buffer 0 ----
    {
        float* nxt = sIn0;
        for (int i = tid; i < (TROWS * NF) / 4; i += 256) {
            const int r  = i >> 7;          // 0..31
            const int k4 = i & 127;         // 0..127
            async_ld16(&nxt[r * SSTR + 4 * k4], &gbase[r * NF + 4 * k4]);
        }
    }

    // ---- stage Wt (transposed bf16), bt, ut : once per block ----
    for (int i = tid; i < (NU * NF) / 4; i += 256) {
        const int k  = i >> 5;              // 0..511
        const int n4 = (i & 31) * 4;        // 0..124
        const float4 w = *(const float4*)&Wt[k * NU + n4];
        sWt[(n4 + 0) * WSTR + k] = (__bf16)w.x;
        sWt[(n4 + 1) * WSTR + k] = (__bf16)w.y;
        sWt[(n4 + 2) * WSTR + k] = (__bf16)w.z;
        sWt[(n4 + 3) * WSTR + k] = (__bf16)w.w;
    }
    if (tid < NU) { sBt[tid] = bt[tid]; sUt[tid] = ut[tid]; }

    // ---- online softmax running state (identical on every thread) ----
    float runM = -INFINITY;
    float runS = 0.f;
    float2 cacc = make_float2(0.f, 0.f);   // context for f = 2*tid, 2*tid+1

    const int n0 = (2 * ng) * 16 + lane16;
    const int n1 = n0 + 16;
    const int mrow  = mt * 16 + lane16;    // tile-local A row for this lane
    const int akoff = laneHi * 16;

    for (int it = 0; it < NTILES; ++it) {
        const int t0  = it * TROWS;
        float*    cur = (it & 1) ? sIn1 : sIn0;
        float*    nxt = (it & 1) ? sIn0 : sIn1;

        // current tile's DMA complete (this wave) + visible to all waves
        wait_async0();
        __syncthreads();

        // ---- kick off async DMA of the NEXT tile (overlaps the compute) ----
        if (it + 1 < NTILES) {
            const float* gn = gbase + (long)(t0 + TROWS) * NF;
            for (int i = tid; i < (TROWS * NF) / 4; i += 256) {
                const int r  = i >> 7;
                const int k4 = i & 127;
                async_ld16(&nxt[r * SSTR + 4 * k4], &gn[r * NF + 4 * k4]);
            }
        }

        // ---- GEMM on cur: 2 WMMAs per wave per K-step ----
        const v8f zero = {0.f,0.f,0.f,0.f,0.f,0.f,0.f,0.f};
        v8f acc0 = zero, acc1 = zero;
#pragma unroll 8
        for (int k0 = 0; k0 < NF; k0 += 32) {
            // A fragment: f32 from LDS tile -> bf16 (v_cvt_pk_bf16_f32)
            const float* ap = &cur[mrow * SSTR + k0 + akoff];
            v16bf a;
#pragma unroll
            for (int j = 0; j < 8; ++j) a[j]     = (__bf16)ap[j];
#pragma unroll
            for (int j = 0; j < 8; ++j) a[8 + j] = (__bf16)ap[8 + j];

            // B fragments: contiguous bf16 in LDS (conflict-free ds_load_b128)
            const int bko = k0 + 16 * laneHi;
            const v8bf b0l = *(const v8bf*)&sWt[n0 * WSTR + bko];
            const v8bf b0h = *(const v8bf*)&sWt[n0 * WSTR + bko + 8];
            const v8bf b1l = *(const v8bf*)&sWt[n1 * WSTR + bko];
            const v8bf b1h = *(const v8bf*)&sWt[n1 * WSTR + bko + 8];
            const v16bf b0 = __builtin_shufflevector(b0l, b0h,
                0,1,2,3,4,5,6,7,8,9,10,11,12,13,14,15);
            const v16bf b1 = __builtin_shufflevector(b1l, b1h,
                0,1,2,3,4,5,6,7,8,9,10,11,12,13,14,15);

            acc0 = __builtin_amdgcn_wmma_f32_16x16x32_bf16(
                false, a, false, b0, (short)0, acc0, false, false);
            acc1 = __builtin_amdgcn_wmma_f32_16x16x32_bf16(
                false, a, false, b1, (short)0, acc1, false, false);
        }

        // ---- epilogue: tanh + ut dot over this wave's two columns ----
        {
            const float bt0 = sBt[n0], ut0 = sUt[n0];
            const float bt1 = sBt[n1], ut1 = sUt[n1];
            float p[8];
#pragma unroll
            for (int r = 0; r < 8; ++r)
                p[r] = tanhf(acc0[r] + bt0) * ut0 + tanhf(acc1[r] + bt1) * ut1;
            // reduce across the 16 lanes sharing each M row
#pragma unroll
            for (int m = 1; m < 16; m <<= 1) {
#pragma unroll
                for (int r = 0; r < 8; ++r)
                    p[r] += __shfl_xor(p[r], m, 32);
            }
            if (lane16 == 0) {
#pragma unroll
                for (int r = 0; r < 8; ++r)
                    sPt[wave * 16 + laneHi * 8 + r] = p[r];
            }
        }
        __syncthreads();

        // ---- deterministic cross-wave combine: 4 N-groups per M row ----
        if (tid < TROWS) {
            const int rmt = tid >> 4;       // row's M-tile
            const int rr  = tid & 15;
            float v = 0.f;
#pragma unroll
            for (int j = 0; j < 4; ++j)
                v += sPt[(2 * j + rmt) * 16 + rr];
            sA2[t0 + tid] = v;
        }
        __syncthreads();

        // ---- online softmax update (all threads, identical scalars) ----
        float tmax = -INFINITY;
#pragma unroll
        for (int r = 0; r < TROWS; ++r) tmax = fmaxf(tmax, sA2[t0 + r]);
        const float newM  = fmaxf(runM, tmax);
        const float scale = __expf(runM - newM);
        cacc.x *= scale; cacc.y *= scale; runS *= scale;
#pragma unroll 4
        for (int r = 0; r < TROWS; ++r) {
            const float  w = __expf(sA2[t0 + r] - newM);
            runS += w;
            const float2 v = *(const float2*)&cur[r * SSTR + 2 * tid];
            cacc.x += w * v.x;
            cacc.y += w * v.y;
        }
        runM = newM;
    }

    // ---- tail: normalize and write outputs ----
    const float inv = 1.f / runS;
    float2 out;
    out.x = cacc.x * inv;
    out.y = cacc.y * inv;
    *(float2*)&ctx[(long)b * NF + 2 * tid] = out;

    tw[b * NT + tid]       = __expf(sA2[tid]       - runM) * inv;
    tw[b * NT + tid + 256] = __expf(sA2[tid + 256] - runM) * inv;
    // feature_weights are identically 1.0 (softmax over a singleton axis)
    fw[b * NT + tid]       = 1.f;
    fw[b * NT + tid + 256] = 1.f;
}

// ---------------------------------------------------------------------------
extern "C" void kernel_launch(void* const* d_in, const int* in_sizes, int n_in,
                              void* d_out, int out_size, void* d_ws, size_t ws_size,
                              hipStream_t stream)
{
    const float* inp = (const float*)d_in[0];
    // d_in[1..3] = Wf, bf, uf — mathematically dead (softmax over singleton axis)
    const float* Wt  = (const float*)d_in[4];
    const float* bt  = (const float*)d_in[5];
    const float* ut  = (const float*)d_in[6];

    float* ctx = (float*)d_out;            // [NB*NF]
    float* fw  = ctx + NB * NF;            // [NT*NF] (all ones)
    float* tw  = fw  + NT * NF;            // [NB*NT]

    fused_attn_kernel<<<NB, 256, SMEM_BYTES, stream>>>(inp, Wt, bt, ut, ctx, fw, tw);

    (void)in_sizes; (void)n_in; (void)out_size; (void)d_ws; (void)ws_size;
}
